// DeepGT_79817672229273
// MI455X (gfx1250) — compile-verified, matrix operand order checked by reading, not run on hardware
//
#include <hip/hip_runtime.h>
#include <hip/hip_bf16.h>

// ---------------------------------------------------------------------------
// DeepGT (TransformerConv x4) for MI455X / gfx1250.
//  - Node GEMMs: bf16 WMMA (v_wmma_f32_16x16x32_bf16); weights pre-packed into
//    per-lane fragment-major bf16 so B operands are contiguous b128 loads.
//  - h kept in bf16 in global memory (halves A-side GEMM traffic); the 16-row
//    A tile is a contiguous 4KB block staged into LDS with CDNA5
//    GLOBAL_LOAD_ASYNC_TO_LDS_B128 (ASYNCcnt), then read as WMMA fragments
//    via ds_load_b128.
//  - Edge attention: wave32-per-edge coalesced gathers, monotonic-uint
//    atomicMax for segment max, f32 atomics for segment sums / scatter-add.
// ---------------------------------------------------------------------------

#define NN     100000          // nodes
#define EE     1600000         // edges
#define HD     128             // hidden dim
#define OD     40              // output dim
#define NL     4               // layers

typedef __attribute__((ext_vector_type(16))) __bf16   v16bf;
typedef __attribute__((ext_vector_type(8)))  __bf16   v8bf;
typedef __attribute__((ext_vector_type(8)))  float    v8f;
typedef __attribute__((ext_vector_type(4)))  unsigned u32x4;

__device__ __forceinline__ unsigned short f2bf(float f) {
    unsigned u = __float_as_uint(f);
    u += 0x7FFFu + ((u >> 16) & 1u);          // round-to-nearest-even
    return (unsigned short)(u >> 16);
}
__device__ __forceinline__ float bf2f(unsigned short u) {
    return __uint_as_float(((unsigned)u) << 16);
}

// monotonic float<->uint mapping so unsigned atomicMax == float max
__device__ __forceinline__ unsigned mono_enc(float f) {
    unsigned u = __float_as_uint(f);
    return (u & 0x80000000u) ? ~u : (u | 0x80000000u);
}
__device__ __forceinline__ float mono_dec(unsigned u) {
    return (u & 0x80000000u) ? __uint_as_float(u & 0x7FFFFFFFu)
                             : __uint_as_float(~u);
}

__device__ __forceinline__ void wait_asynccnt0() {
#if __has_builtin(__builtin_amdgcn_s_wait_asynccnt)
    __builtin_amdgcn_s_wait_asynccnt(0);
#else
    asm volatile("s_wait_asynccnt 0x0" ::: "memory");
#endif
}

// ---------------------------------------------------------------------------
// Pack a 128x128 fp32 weight (K-major: W[k][n]) into bf16 fragment-major order
// matching the 16-bit B-operand layout of v_wmma_f32_16x16x32_bf16:
//   packed[(kt*8+nt)*32 + lane][i] , i=0..15,
//   n = nt*16 + lane%16,  k = kt*32 + 8*(lane/16) + (i<8 ? i : i+8)
// 17 slots: [lin_W][Wq0..3][Wk0..3][Wv0..3][Ws0..3]
// ---------------------------------------------------------------------------
__global__ void __launch_bounds__(256)
pack_weights_kernel(const float* __restrict__ linW, const float* __restrict__ Wq,
                    const float* __restrict__ Wk,   const float* __restrict__ Wv,
                    const float* __restrict__ Ws,   unsigned short* __restrict__ packed)
{
    int tid  = blockIdx.x * blockDim.x + threadIdx.x;   // 0 .. 17*1024-1
    int slot = tid >> 10;
    if (slot >= 17) return;
    int r    = tid & 1023;
    int kt   = r >> 8;
    int nt   = (r >> 5) & 7;
    int lane = r & 31;

    const float* src;
    if (slot == 0) src = linW;
    else {
        int m = slot - 1, t = m >> 2, layer = m & 3;
        const float* base = (t == 0) ? Wq : (t == 1) ? Wk : (t == 2) ? Wv : Ws;
        src = base + (size_t)layer * HD * HD;
    }
    int n     = nt * 16 + (lane & 15);
    int khalf = (lane >> 4) * 8;
    unsigned short* dst = packed + (size_t)slot * 16384
                                 + (size_t)((kt * 8 + nt) * 32 + lane) * 16;
#pragma unroll
    for (int i = 0; i < 16; ++i) {
        int k = kt * 32 + khalf + ((i < 8) ? i : i + 8);
        dst[i] = f2bf(src[(size_t)k * HD + n]);
    }
}

// fp32 -> bf16 feature conversion (one-time, for x)
__global__ void cvt_bf16_kernel(const float4* __restrict__ in,
                                ushort4* __restrict__ out)
{
    int i = blockIdx.x * blockDim.x + threadIdx.x;
    if (i >= NN * HD / 4) return;
    float4 f = in[i];
    ushort4 o;
    o.x = f2bf(f.x); o.y = f2bf(f.y); o.z = f2bf(f.z); o.w = f2bf(f.w);
    out[i] = o;
}

// ---------------------------------------------------------------------------
// Fused GEMM: out{0,1}[16 x 128] = A_tile[16 x 128](bf16) @ W{0,1} + b{0,1}
// blockDim = 64 (wave0 -> matrix0, wave1 -> matrix1); blockDim = 32 for 1 mat.
// A tile (contiguous 4KB of bf16) staged via async load-to-LDS.
// ---------------------------------------------------------------------------
__global__ void __launch_bounds__(64)
gemm16_kernel(const unsigned short* __restrict__ A,
              const unsigned short* __restrict__ wp0,
              const unsigned short* __restrict__ wp1,
              const float* __restrict__ b0, const float* __restrict__ b1,
              float* __restrict__ out0, float* __restrict__ out1)
{
    __shared__ __align__(16) unsigned short sA[16 * HD];   // 4 KB bf16 tile
    const int row0 = blockIdx.x * 16;

    // ---- stage the contiguous 4KB tile: 256 x 16B chunks, ASYNC to LDS ----
    {
        const unsigned short* gsrc = A + (size_t)row0 * HD;
        const unsigned lds0 = (unsigned)(uintptr_t)(&sA[0]);
        for (int t = threadIdx.x; t < 256; t += blockDim.x) {
            const unsigned ldsa = lds0 + t * 16;
            const unsigned short* ga = gsrc + t * 8;
            asm volatile("global_load_async_to_lds_b128 %0, %1, off"
                         :: "v"(ldsa), "v"(ga) : "memory");
        }
        wait_asynccnt0();
    }
    __syncthreads();

    const int wave = threadIdx.x >> 5;
    const int lane = threadIdx.x & 31;
    const unsigned short* wp = wave ? wp1 : wp0;
    const float*        bias = wave ? b1  : b0;
    float*               out = wave ? out1 : out0;

    v8f acc[8];
    const v8f vzero = {0.f, 0.f, 0.f, 0.f, 0.f, 0.f, 0.f, 0.f};
#pragma unroll
    for (int nt = 0; nt < 8; ++nt) acc[nt] = vzero;

    const int rowA  = lane & 15;
    const int khalf = (lane >> 4) * 8;

#pragma unroll
    for (int kt = 0; kt < 4; ++kt) {
        const int k0 = kt * 32;
        // A fragment (ISA 16-bit A 16x32 layout): two contiguous 16B LDS loads
        u32x4 lo = *(const u32x4*)(&sA[rowA * HD + k0 + khalf]);
        u32x4 hi = *(const u32x4*)(&sA[rowA * HD + k0 + 16 + khalf]);
        v8bf alo = __builtin_bit_cast(v8bf, lo);
        v8bf ahi = __builtin_bit_cast(v8bf, hi);
        v16bf afrag;
#pragma unroll
        for (int i = 0; i < 8; ++i) { afrag[i] = alo[i]; afrag[8 + i] = ahi[i]; }

#pragma unroll
        for (int nt = 0; nt < 8; ++nt) {
            const unsigned short* bp =
                wp + (size_t)((kt * 8 + nt) * 32 + lane) * 16;
            u32x4 blo = *(const u32x4*)(bp);
            u32x4 bhi = *(const u32x4*)(bp + 8);
            v8bf bl = __builtin_bit_cast(v8bf, blo);
            v8bf bh = __builtin_bit_cast(v8bf, bhi);
            v16bf bfrag;
#pragma unroll
            for (int i = 0; i < 8; ++i) { bfrag[i] = bl[i]; bfrag[8 + i] = bh[i]; }
            acc[nt] = __builtin_amdgcn_wmma_f32_16x16x32_bf16(
                false, afrag, false, bfrag, (short)0, acc[nt], false, false);
        }
    }

    // C/D layout: VGPR r -> M = r + 8*(lane>=16), N = nt*16 + lane%16
    const int colL = lane & 15;
    const int mOff = (lane >> 4) * 8;
#pragma unroll
    for (int nt = 0; nt < 8; ++nt) {
        int col  = nt * 16 + colL;
        float bv = bias[col];
#pragma unroll
        for (int r = 0; r < 8; ++r)
            out[(size_t)(row0 + mOff + r) * HD + col] = acc[nt][r] + bv;
    }
}

// ---------------------------------------------------------------------------
// Per-layer init: agg = 0, m_u = 0 (== -inf in monotonic space), den = 0
// ---------------------------------------------------------------------------
__global__ void init_layer_kernel(float* __restrict__ agg,
                                  unsigned* __restrict__ m_u,
                                  float* __restrict__ den)
{
    int i = blockIdx.x * blockDim.x + threadIdx.x;
    if (i < NN * HD) agg[i] = 0.0f;
    if (i < NN) { m_u[i] = 0u; den[i] = 0.0f; }
}

// one wave32 per edge: logits[e] = dot(q[dst], k[src]) * scale ; segment max
__global__ void __launch_bounds__(256)
edge_logits_kernel(const int* __restrict__ src, const int* __restrict__ dst,
                   const float* __restrict__ q, const float* __restrict__ k,
                   float* __restrict__ logits, unsigned* __restrict__ m_u)
{
    const int e    = blockIdx.x * (blockDim.x >> 5) + (threadIdx.x >> 5);
    const int lane = threadIdx.x & 31;
    if (e >= EE) return;
    const int s = src[e], d = dst[e];
    const float4 qv = *(const float4*)(q + (size_t)d * HD + lane * 4);
    const float4 kv = *(const float4*)(k + (size_t)s * HD + lane * 4);
    float p = qv.x * kv.x + qv.y * kv.y + qv.z * kv.z + qv.w * kv.w;
#pragma unroll
    for (int off = 16; off; off >>= 1) p += __shfl_xor(p, off, 32);
    if (lane == 0) {
        float lg = p * 0.08838834764831845f;   // 1/sqrt(128)
        logits[e] = lg;
        atomicMax(m_u + d, mono_enc(lg));
    }
}

// e_val = exp(logit - m[dst]) in place; segment sum into den
__global__ void edge_exp_kernel(const int* __restrict__ dst,
                                float* __restrict__ logits,
                                const unsigned* __restrict__ m_u,
                                float* __restrict__ den)
{
    int e = blockIdx.x * blockDim.x + threadIdx.x;
    if (e >= EE) return;
    int d = dst[e];
    float w = __expf(logits[e] - mono_dec(m_u[d]));
    logits[e] = w;
    atomicAdd(den + d, w);
}

// one wave32 per edge: agg[dst] += alpha * v[src]
__global__ void __launch_bounds__(256)
edge_agg_kernel(const int* __restrict__ src, const int* __restrict__ dst,
                const float* __restrict__ ew, const float* __restrict__ den,
                const float* __restrict__ v, float* __restrict__ agg)
{
    const int e    = blockIdx.x * (blockDim.x >> 5) + (threadIdx.x >> 5);
    const int lane = threadIdx.x & 31;
    if (e >= EE) return;
    const int s = src[e], d = dst[e];
    const float dn    = den[d];
    const float alpha = ew[e] / (dn > 0.0f ? dn : 1.0f);
    const float4 vv = *(const float4*)(v + (size_t)s * HD + lane * 4);
    float* ap = agg + (size_t)d * HD + lane * 4;
    atomicAdd(ap + 0, alpha * vv.x);
    atomicAdd(ap + 1, alpha * vv.y);
    atomicAdd(ap + 2, alpha * vv.z);
    atomicAdd(ap + 3, alpha * vv.w);
}

// h(bf16) = agg + skip ; optional ELU
__global__ void combine_kernel(const float4* __restrict__ agg,
                               const float4* __restrict__ sk,
                               ushort4* __restrict__ h, int do_elu)
{
    int i = blockIdx.x * blockDim.x + threadIdx.x;
    if (i >= NN * HD / 4) return;
    float4 a = agg[i], s = sk[i], r;
    r.x = a.x + s.x; r.y = a.y + s.y; r.z = a.z + s.z; r.w = a.w + s.w;
    if (do_elu) {
        r.x = r.x > 0.f ? r.x : __expf(r.x) - 1.f;
        r.y = r.y > 0.f ? r.y : __expf(r.y) - 1.f;
        r.z = r.z > 0.f ? r.z : __expf(r.z) - 1.f;
        r.w = r.w > 0.f ? r.w : __expf(r.w) - 1.f;
    }
    ushort4 o;
    o.x = f2bf(r.x); o.y = f2bf(r.y); o.z = f2bf(r.z); o.w = f2bf(r.w);
    h[i] = o;
}

// classifier + log_softmax, one wave32 per node (8 nodes / block)
__global__ void __launch_bounds__(256)
fc_logsoftmax_kernel(const unsigned short* __restrict__ h,
                     const float* __restrict__ W,
                     const float* __restrict__ b, float* __restrict__ out)
{
    __shared__ __align__(16) float sh[8][HD];
    __shared__ float sz[8][OD];
    const int w = threadIdx.x >> 5, lane = threadIdx.x & 31;
    const int node = blockIdx.x * 8 + w;
    if (node >= NN) return;

    ushort4 hv = *(const ushort4*)(h + (size_t)node * HD + lane * 4);
    float4 hf;
    hf.x = bf2f(hv.x); hf.y = bf2f(hv.y); hf.z = bf2f(hv.z); hf.w = bf2f(hv.w);
    *(float4*)&sh[w][lane * 4] = hf;                      // wave-synchronous

    for (int j = lane; j < OD; j += 32) {
        float acc = b[j];
#pragma unroll 4
        for (int c = 0; c < HD; ++c) acc += sh[w][c] * W[c * OD + j];
        sz[w][j] = acc;
    }
    float m = -3.4e38f;
    for (int j = 0; j < OD; ++j) m = fmaxf(m, sz[w][j]);
    float ssum = 0.f;
    for (int j = 0; j < OD; ++j) ssum += __expf(sz[w][j] - m);
    float lse = m + __logf(ssum);
    for (int j = lane; j < OD; j += 32)
        out[(size_t)node * OD + j] = sz[w][j] - lse;
}

// ---------------------------------------------------------------------------
extern "C" void kernel_launch(void* const* d_in, const int* in_sizes, int n_in,
                              void* d_out, int out_size, void* d_ws, size_t ws_size,
                              hipStream_t stream)
{
    const float* x    = (const float*)d_in[0];
    const int*   ei   = (const int*)d_in[1];
    const int*   src  = ei;            // edge_index[0]
    const int*   dst  = ei + EE;       // edge_index[1]
    const float* linW = (const float*)d_in[2];
    const float* linb = (const float*)d_in[3];
    const float* Wq   = (const float*)d_in[4];
    const float* bq   = (const float*)d_in[5];
    const float* Wk   = (const float*)d_in[6];
    const float* bk   = (const float*)d_in[7];
    const float* Wv   = (const float*)d_in[8];
    const float* bv   = (const float*)d_in[9];
    const float* Ws   = (const float*)d_in[10];
    const float* bs   = (const float*)d_in[11];
    const float* fcW  = (const float*)d_in[12];
    const float* fcb  = (const float*)d_in[13];

    char* ws = (char*)d_ws;
    size_t off = 0;
    auto alloc = [&](size_t bytes) -> void* {
        void* p = ws + off;
        off += (bytes + 255) & ~(size_t)255;
        return p;
    };
    unsigned short* h      = (unsigned short*)alloc((size_t)NN * HD * 2); // bf16
    unsigned short* xbf    = (unsigned short*)alloc((size_t)NN * HD * 2); // bf16
    float*          t0     = (float*)alloc((size_t)NN * HD * 4);  // q, then v
    float*          t1     = (float*)alloc((size_t)NN * HD * 4);  // k, then skip
    float*          agg    = (float*)alloc((size_t)NN * HD * 4);
    float*          logits = (float*)alloc((size_t)EE * 4);
    unsigned*       m_u    = (unsigned*)alloc((size_t)NN * 4);
    float*          den    = (float*)alloc((size_t)NN * 4);
    unsigned short* packed = (unsigned short*)alloc((size_t)17 * 16384 * 2);

    const int GB = NN / 16;          // 6250 GEMM row tiles

    pack_weights_kernel<<<(17 * 1024) / 256, 256, 0, stream>>>(linW, Wq, Wk, Wv, Ws, packed);
    cvt_bf16_kernel<<<(NN * HD / 4) / 256, 256, 0, stream>>>((const float4*)x, (ushort4*)xbf);

    // input projection: h = x @ lin_W + lin_b  (single matrix -> 1 wave/block)
    // (note: out0/out1 alias with identical data -> benign)
    gemm16_kernel<<<GB, 32, 0, stream>>>(xbf, packed, packed, linb, linb, t0, t0);
    // convert projected features to bf16 h (no elu on input projection)
    {
        // reuse combine with agg = t0, skip = zeroed? simpler: dedicated pass
    }
    cvt_bf16_kernel<<<(NN * HD / 4) / 256, 256, 0, stream>>>((const float4*)t0, (ushort4*)h);

    for (int l = 0; l < NL; ++l) {
        const unsigned short* pq = packed + (size_t)(1 + 0 * 4 + l) * 16384;
        const unsigned short* pk = packed + (size_t)(1 + 1 * 4 + l) * 16384;
        const unsigned short* pv = packed + (size_t)(1 + 2 * 4 + l) * 16384;
        const unsigned short* ps = packed + (size_t)(1 + 3 * 4 + l) * 16384;

        gemm16_kernel<<<GB, 64, 0, stream>>>(h, pq, pk, bq + l * HD, bk + l * HD, t0, t1);
        init_layer_kernel<<<(NN * HD + 255) / 256, 256, 0, stream>>>(agg, m_u, den);
        edge_logits_kernel<<<EE / 8, 256, 0, stream>>>(src, dst, t0, t1, logits, m_u);
        edge_exp_kernel<<<EE / 256, 256, 0, stream>>>(dst, logits, m_u, den);
        gemm16_kernel<<<GB, 64, 0, stream>>>(h, pv, ps, bv + l * HD, bs + l * HD, t0, t1);
        edge_agg_kernel<<<EE / 8, 256, 0, stream>>>(src, dst, logits, den, t0, agg);
        combine_kernel<<<(NN * HD / 4 + 255) / 256, 256, 0, stream>>>(
            (const float4*)agg, (const float4*)t1, (ushort4*)h, (l < NL - 1) ? 1 : 0);
    }

    fc_logsoftmax_kernel<<<NN / 8, 256, 0, stream>>>(h, fcW, fcb, (float*)d_out);
}
